// AttentionBlock_90537910600269
// MI455X (gfx1250) — matrix operand order repaired and checked by reading, not compiled, not run on hardware
//
#include <hip/hip_runtime.h>

// CDNA5 / gfx1250, wave32. HBM-bandwidth-bound gating kernel:
//   out[p] = sigmoid(leaky(xa+xs,0.3)*w2 + b2) * xs
//   xa = xatt[p,:] @ W_act, xs = xsaut[p,:] @ W_saut   (C = 64)
// Dot products via v_wmma_f32_16x16x32_bf16; 2 tiles of 16 positions per wave.
// Weights pre-converted to bf16 once (prepass) into d_ws; A-operands packed to
// bf16 with single v_perm_b32 per pair (truncation) to stay under issue-rate limits.

typedef __attribute__((ext_vector_type(16))) __bf16       v16bf;
typedef __attribute__((ext_vector_type(8)))  float        v8f;
typedef __attribute__((ext_vector_type(4)))  float        v4f;
typedef __attribute__((ext_vector_type(8)))  unsigned int v8u;

// ---- prepass helper: proper RNE f32 -> bf16 (runs once on 128 elements) ----
static __device__ __forceinline__ __bf16 f2bf_rne(float f) {
    unsigned int u = __float_as_uint(f);
    u += 0x7FFFu + ((u >> 16) & 1u);
    unsigned short h = (unsigned short)(u >> 16);
    return __builtin_bit_cast(__bf16, h);
}

__global__ void convert_weights_kernel(const float* __restrict__ Wact,
                                       const float* __restrict__ Wsaut,
                                       __bf16* __restrict__ wsb) {
    const int i = threadIdx.x;          // launched with 64 threads
    wsb[i]      = f2bf_rne(Wact[i]);
    wsb[64 + i] = f2bf_rne(Wsaut[i]);
}

// ---- main kernel helpers ----

// pack two f32 into packed bf16x2 (truncate) with a single v_perm_b32
static __device__ __forceinline__ unsigned int pack_bf16x2(float lo, float hi) {
    // dst bytes: {hi.b3, hi.b2, lo.b3, lo.b2}; {S0,S1}={hi,lo}: b's bytes are 0..3, a's are 4..7
    return __builtin_amdgcn_perm(__float_as_uint(hi), __float_as_uint(lo), 0x07060302u);
}

// Load one 64-float row's four 8-float chunks for this lane-half (contiguous k
// order k = 16*hi + s, same as the B fragments) and pack into two v16bf A fragments (K=0..31, K=32..63).
static __device__ __forceinline__ void load_row(const float* __restrict__ row, int b,
                                                v16bf& a1, v16bf& a2) {
    const v4f* q = (const v4f*)row;
    v4f c0a = __builtin_nontemporal_load(q + b + 0);   // K = 16*hi + 0..3
    v4f c0b = __builtin_nontemporal_load(q + b + 1);   // K = 16*hi + 4..7
    v4f c1a = __builtin_nontemporal_load(q + b + 2);   // K = 16*hi + 8..11
    v4f c1b = __builtin_nontemporal_load(q + b + 3);
    v4f c2a = __builtin_nontemporal_load(q + b + 8);   // K = 32 + 16*hi + 0..3
    v4f c2b = __builtin_nontemporal_load(q + b + 9);
    v4f c3a = __builtin_nontemporal_load(q + b + 10);  // K = 32 + 16*hi + 8..11
    v4f c3b = __builtin_nontemporal_load(q + b + 11);
    v8u p1, p2;
    p1[0] = pack_bf16x2(c0a[0], c0a[1]);
    p1[1] = pack_bf16x2(c0a[2], c0a[3]);
    p1[2] = pack_bf16x2(c0b[0], c0b[1]);
    p1[3] = pack_bf16x2(c0b[2], c0b[3]);
    p1[4] = pack_bf16x2(c1a[0], c1a[1]);
    p1[5] = pack_bf16x2(c1a[2], c1a[3]);
    p1[6] = pack_bf16x2(c1b[0], c1b[1]);
    p1[7] = pack_bf16x2(c1b[2], c1b[3]);
    p2[0] = pack_bf16x2(c2a[0], c2a[1]);
    p2[1] = pack_bf16x2(c2a[2], c2a[3]);
    p2[2] = pack_bf16x2(c2b[0], c2b[1]);
    p2[3] = pack_bf16x2(c2b[2], c2b[3]);
    p2[4] = pack_bf16x2(c3a[0], c3a[1]);
    p2[5] = pack_bf16x2(c3a[2], c3a[3]);
    p2[6] = pack_bf16x2(c3b[0], c3b[1]);
    p2[7] = pack_bf16x2(c3b[2], c3b[3]);
    a1 = __builtin_bit_cast(v16bf, p1);
    a2 = __builtin_bit_cast(v16bf, p2);
}

__global__ __launch_bounds__(256) void AttentionBlock_gate_kernel(
    const float* __restrict__ xatt,  const float* __restrict__ xsaut,
    const __bf16* __restrict__ wsb,
    const float* __restrict__ bact,  const float* __restrict__ bsaut,
    const float* __restrict__ W2p,   const float* __restrict__ b2p,
    float* __restrict__ out, int nPos) {
    const int lane  = threadIdx.x & 31;
    const int wid   = (int)((blockIdx.x * blockDim.x + threadIdx.x) >> 5);
    const int base0 = wid * 32;                  // 2 tiles of 16 positions per wave
    if (base0 >= nPos) return;                   // wave-uniform

    const int m  = lane & 15;
    const int hi = lane >> 4;
    const int b  = hi * 4;

    // B fragments: bf16 weights pre-converted by prepass; replicated across all
    // 16 N-columns by construction of the layout (lanes 0-15: K=0..15, 16-31: K=16..31).
    const __bf16* wA = wsb + hi * 16;
    const __bf16* wS = wsb + 64 + hi * 16;
    v16bf bA1 = *(const v16bf*)(wA);
    v16bf bA2 = *(const v16bf*)(wA + 32);
    v16bf bS1 = *(const v16bf*)(wS);
    v16bf bS2 = *(const v16bf*)(wS + 32);

    const float ba  = bact[0];
    const float bs_ = bsaut[0];
    const float w2  = W2p[0];
    const float b2v = b2p[0];

#pragma unroll
    for (int t = 0; t < 2; ++t) {
        const int base = base0 + t * 16;
        if (base >= nPos) break;                 // wave-uniform

        const float* rowA = xatt  + (size_t)(base + m) * 64;
        const float* rowS = xsaut + (size_t)(base + m) * 64;

        v16bf aa1, aa2, as1, as2;
        load_row(rowA, b, aa1, aa2);
        load_row(rowS, b, as1, as2);

        v8f ca = {0.f,0.f,0.f,0.f,0.f,0.f,0.f,0.f};
        v8f cs = {0.f,0.f,0.f,0.f,0.f,0.f,0.f,0.f};
        ca = __builtin_amdgcn_wmma_f32_16x16x32_bf16(false, aa1, false, bA1, (short)0, ca, false, false);
        ca = __builtin_amdgcn_wmma_f32_16x16x32_bf16(false, aa2, false, bA2, (short)0, ca, false, false);
        cs = __builtin_amdgcn_wmma_f32_16x16x32_bf16(false, as1, false, bS1, (short)0, cs, false, false);
        cs = __builtin_amdgcn_wmma_f32_16x16x32_bf16(false, as2, false, bS2, (short)0, cs, false, false);

        // Every lane's v8f holds rows 8*hi..8*hi+7 (all columns identical).
        v4f o_lo, o_hi;
#pragma unroll
        for (int r = 0; r < 8; ++r) {
            float xa = ca[r] + ba;
            float xs = cs[r] + bs_;
            float h  = xa + xs;
            h = (h >= 0.f) ? h : 0.3f * h;                    // LeakyReLU(0.3)
            float g = 1.f / (1.f + __expf(-(h * w2 + b2v)));  // sigmoid
            float o = g * xs;
            if (r < 4) o_lo[r] = o; else o_hi[r - 4] = o;
        }
        if (m == 0) {  // lanes 0 and 16 write the tile's 16 contiguous outputs
            v4f* op = (v4f*)(out + base + hi * 8);
            __builtin_nontemporal_store(o_lo, op);
            __builtin_nontemporal_store(o_hi, op + 1);
        }
        // EXEC re-converges here before the next iteration's WMMAs.
    }
}

extern "C" void kernel_launch(void* const* d_in, const int* in_sizes, int n_in,
                              void* d_out, int out_size, void* d_ws, size_t ws_size,
                              hipStream_t stream) {
    const float* xatt  = (const float*)d_in[0];
    const float* xsaut = (const float*)d_in[1];
    const float* Wact  = (const float*)d_in[2];
    const float* bact  = (const float*)d_in[3];
    const float* Wsaut = (const float*)d_in[4];
    const float* bsaut = (const float*)d_in[5];
    const float* W2p   = (const float*)d_in[6];
    const float* b2p   = (const float*)d_in[7];
    float* out   = (float*)d_out;
    __bf16* wsb  = (__bf16*)d_ws;    // 128 bf16 = 256 bytes of scratch

    // Prepass: convert the two 64-entry weight vectors to bf16 once.
    convert_weights_kernel<<<1, 64, 0, stream>>>(Wact, Wsaut, wsb);

    const int nPos   = in_sizes[0] / 64;              // B*L positions
    const int nTiles = (nPos + 15) / 16;
    const int nWaves = (nTiles + 1) / 2;              // 2 tiles per wave
    const int threads = 256;                          // 8 wave32s per block
    const long long totalThreads = (long long)nWaves * 32;
    const int blocks = (int)((totalThreads + threads - 1) / threads);

    AttentionBlock_gate_kernel<<<blocks, threads, 0, stream>>>(
        xatt, xsaut, wsb, bact, bsaut, W2p, b2p, out, nPos);
}
